// Generator_41377714930225
// MI455X (gfx1250) — matrix operand order repaired and checked by reading, not compiled, not run on hardware
//
#include <hip/hip_runtime.h>

// ---------------------------------------------------------------------------
// ModLinear forward for MI455X (gfx1250, wave32, WMMA).
//   alpha = z @ Wa^T + ba        [2,512]   (tiny)
//   beta  = z @ Wb^T + bb        [2,512]   (tiny)
//   out   = x @ (W*diag(alpha))^T + beta   [2,131072,512]  <- 137 GFLOP main GEMM
// Main GEMM uses bf16 hi/lo split (3x v_wmma_f32_16x16x32_bf16, f32 accum):
// ~2^-16 relative error, ~80us compute vs 46us HBM floor on MI455X.
// Modulated weight is pre-packed into per-lane WMMA B-fragment layout (bf16
// hi + lo residual, 2 MB total, L2-resident).
// ---------------------------------------------------------------------------

typedef __attribute__((ext_vector_type(16))) __bf16       v16bf;
typedef __attribute__((ext_vector_type(8)))  float        v8f;
typedef __attribute__((ext_vector_type(4)))  float        v4f;
typedef __attribute__((ext_vector_type(8)))  unsigned int v8u;

union BfFrag { v8u d; v16bf b; };

#define IN_F    512
#define OUT_F   512
#define STYLE_F 256
#define NROW    131072   // rows per batch; total rows = 2*131072

// Pack two f32 -> one dword of two truncated bf16 (low16 = f0, high16 = f1).
// v_perm_b32: single instruction grabs top halves of both sources.
__device__ static inline unsigned pk(float f0, float f1) {
  return __builtin_amdgcn_perm(__float_as_uint(f1), __float_as_uint(f0), 0x07060302u);
}
// Residual after bf16 truncation (captured by the lo-term WMMAs).
__device__ static inline float res(float f) {
  return f - __uint_as_float(__float_as_uint(f) & 0xFFFF0000u);
}

// ---------------------------------------------------------------------------
// Kernel 1: alpha[2,512] and beta[2,512] from style vector z[2,256].
// 2048 threads, one output each, K=256 dot product. Negligible cost.
// ---------------------------------------------------------------------------
__global__ __launch_bounds__(256) void modlin_style(
    const float* __restrict__ z,
    const float* __restrict__ wa, const float* __restrict__ ba,
    const float* __restrict__ wb, const float* __restrict__ bb,
    float* __restrict__ alpha, float* __restrict__ beta)
{
  const int t     = blockIdx.x * 256 + threadIdx.x;   // 0..2047
  const int which = t >> 10;                          // 0: alpha, 1: beta
  const int b     = (t >> 9) & 1;
  const int i     = t & 511;
  const float* zr = z + b * STYLE_F;
  const float* w  = (which ? wb : wa) + (size_t)i * STYLE_F;
  float s = which ? bb[i] : ba[i];
  for (int k = 0; k < STYLE_F; k += 4)
    s += zr[k]*w[k] + zr[k+1]*w[k+1] + zr[k+2]*w[k+2] + zr[k+3]*w[k+3];
  (which ? beta : alpha)[b * 512 + i] = s;
}

// ---------------------------------------------------------------------------
// Kernel 2: pack w_mod[b][o][k] = weight[o][k]*alpha[b][k] into per-lane WMMA
// B-fragment layout (bf16 hi + bf16 lo residual).
// Fragment f = ((b*32 + ntile)*16 + kstep); each lane owns 8 contiguous dwords.
// B layout (32x16 bf16): lanes 0-15 hold K=0..7 & 16..23 (2 per VGPR, col N = lane),
// lanes 16-31 hold K=8..15 & 24..31 (col N = lane-16).
// Output dword index == global thread id by construction.
// ---------------------------------------------------------------------------
__global__ __launch_bounds__(256) void modlin_pack(
    const float* __restrict__ weight, const float* __restrict__ alpha,
    unsigned* __restrict__ bhi, unsigned* __restrict__ blo)
{
  const unsigned t    = blockIdx.x * 256 + threadIdx.x;  // 0..262143
  const unsigned d    = t & 7;
  const unsigned lane = (t >> 3) & 31;
  const unsigned ks   = (t >> 8) & 15;
  const unsigned nt   = (t >> 12) & 31;
  const unsigned b    = (t >> 17) & 1;

  const unsigned o     = nt * 16 + (lane & 15);
  const unsigned khalf = (lane < 16) ? 0u : 8u;
  const unsigned k0    = ks * 32 + khalf + 2 * d + ((d >= 4) ? 8u : 0u);

  const float* wr = weight + (size_t)o * IN_F;
  const float* ar = alpha + b * 512;
  const float w0 = wr[k0]     * ar[k0];
  const float w1 = wr[k0 + 1] * ar[k0 + 1];

  const unsigned h0 = __float_as_uint(w0) & 0xFFFF0000u;
  const unsigned h1 = __float_as_uint(w1) & 0xFFFF0000u;
  const float    r0 = w0 - __uint_as_float(h0);
  const float    r1 = w1 - __uint_as_float(h1);

  bhi[t] = h1 | (h0 >> 16);
  blo[t] = (__float_as_uint(r1) & 0xFFFF0000u) | ((__float_as_uint(r0) & 0xFFFF0000u) >> 16);
}

// ---------------------------------------------------------------------------
// Kernel 3: main GEMM. Grid = 16384 M-tiles (16 rows each), block = 256 = 8
// waves; wave w computes its M-tile x out-cols [64w, 64w+64) -> 4 accumulators.
// K loop: 16 steps of 32; per step: build bf16 hi/lo A-fragments in-register
// from fp32 x, then 4 x 3 WMMAs against L2-resident packed B-fragments.
// A layout (16x32 bf16): lanes 0-15 row M=lane, K 0..7 (v0-3) & 16..23 (v4-7);
// lanes 16-31 row M=lane-16, K 8..15 & 24..31.
// ---------------------------------------------------------------------------
__global__ __launch_bounds__(256) void modlin_gemm(
    const float* __restrict__ x,
    const unsigned* __restrict__ bhi, const unsigned* __restrict__ blo,
    const float* __restrict__ beta, float* __restrict__ out)
{
  const int  lane    = threadIdx.x & 31;
  const int  wave    = threadIdx.x >> 5;
  const long rowbase = (long)blockIdx.x * 16;
  const int  b       = (int)(rowbase >> 17);          // rowbase / 131072
  const int  colbase = wave * 64;
  const int  koff    = (lane < 16) ? 0 : 8;
  const float* xrow  = x + (rowbase + (lane & 15)) * (long)IN_F;
  const v16bf* Bh    = (const v16bf*)bhi;
  const v16bf* Bl    = (const v16bf*)blo;

  v8f acc[4];
#pragma unroll
  for (int nt = 0; nt < 4; ++nt)
#pragma unroll
    for (int e = 0; e < 8; ++e) acc[nt][e] = 0.0f;

  for (int ks = 0; ks < 16; ++ks) {
    const float* p = xrow + ks * 32 + koff;
    const v4f q0 = *(const v4f*)(p);
    const v4f q1 = *(const v4f*)(p + 4);
    const v4f q2 = *(const v4f*)(p + 16);
    const v4f q3 = *(const v4f*)(p + 20);

    BfFrag ah, al;
    const v8u hi = { pk(q0.x, q0.y), pk(q0.z, q0.w), pk(q1.x, q1.y), pk(q1.z, q1.w),
                     pk(q2.x, q2.y), pk(q2.z, q2.w), pk(q3.x, q3.y), pk(q3.z, q3.w) };
    const v8u lo = { pk(res(q0.x), res(q0.y)), pk(res(q0.z), res(q0.w)),
                     pk(res(q1.x), res(q1.y)), pk(res(q1.z), res(q1.w)),
                     pk(res(q2.x), res(q2.y)), pk(res(q2.z), res(q2.w)),
                     pk(res(q3.x), res(q3.y)), pk(res(q3.z), res(q3.w)) };
    ah.d = hi; al.d = lo;

    const int fbase = ((b * 32 + (colbase >> 4)) * 16 + ks) * 32 + lane;
#pragma unroll
    for (int nt = 0; nt < 4; ++nt) {
      const v16bf bh = Bh[fbase + nt * 512];
      const v16bf bl = Bl[fbase + nt * 512];
      acc[nt] = __builtin_amdgcn_wmma_f32_16x16x32_bf16(
          false, ah.b, false, bh, (short)0, acc[nt], false, false);
      acc[nt] = __builtin_amdgcn_wmma_f32_16x16x32_bf16(
          false, al.b, false, bh, (short)0, acc[nt], false, false);
      acc[nt] = __builtin_amdgcn_wmma_f32_16x16x32_bf16(
          false, ah.b, false, bl, (short)0, acc[nt], false, false);
    }
  }

  // Epilogue: C/D layout -> VGPR r: lanes 0-15 row M=r, lanes 16-31 row M=r+8.
  const int rlo = (lane < 16) ? 0 : 8;
#pragma unroll
  for (int nt = 0; nt < 4; ++nt) {
    const int   col = colbase + nt * 16 + (lane & 15);
    const float bv  = beta[b * OUT_F + col];
    float*      o   = out + rowbase * OUT_F + col;
#pragma unroll
    for (int r = 0; r < 8; ++r)
      o[(long)(r + rlo) * OUT_F] = acc[nt][r] + bv;
  }
}

// ---------------------------------------------------------------------------
// Host launch. Inputs (setup_inputs order):
//   0:x [2,131072,512] f32   1:z [2,256] f32       2:weight [512,512] f32
//   3:weight_alpha [512,256] 4:bias_alpha [512]    5:weight_beta [512,256]
//   6:bias_beta [512]
// Workspace layout: alpha 4KB | beta 4KB | Bfrag_hi 1MB | Bfrag_lo 1MB.
// ---------------------------------------------------------------------------
extern "C" void kernel_launch(void* const* d_in, const int* in_sizes, int n_in,
                              void* d_out, int out_size, void* d_ws, size_t ws_size,
                              hipStream_t stream) {
  const float* x      = (const float*)d_in[0];
  const float* z      = (const float*)d_in[1];
  const float* weight = (const float*)d_in[2];
  const float* wa     = (const float*)d_in[3];
  const float* ba     = (const float*)d_in[4];
  const float* wb     = (const float*)d_in[5];
  const float* bb     = (const float*)d_in[6];
  float* out = (float*)d_out;

  char* ws = (char*)d_ws;
  float*    alpha = (float*)ws;                         // 2*512 f32
  float*    beta  = (float*)(ws + 4096);                // 2*512 f32
  unsigned* bhi   = (unsigned*)(ws + 8192);             // 1 MB packed bf16 hi
  unsigned* blo   = (unsigned*)(ws + 8192 + (1u << 20));// 1 MB packed bf16 lo

  modlin_style<<<8, 256, 0, stream>>>(z, wa, ba, wb, bb, alpha, beta);
  modlin_pack<<<1024, 256, 0, stream>>>(weight, alpha, bhi, blo);
  modlin_gemm<<<16384, 256, 0, stream>>>(x, bhi, blo, beta, out);
}
